// AdditiveAttention_30348238913921
// MI455X (gfx1250) — matrix operand order repaired and checked by reading, not compiled, not run on hardware
//
#include <hip/hip_runtime.h>
#include <math.h>

#define B_    64
#define T_    4096
#define HID_  1024
#define MEM_  1024
#define ATTN_ 256

#define TM    128            // t-rows per block (8 waves x 16 rows)
#define KC    64             // K chunk staged in LDS
#define BPAD  16             // bf16 padding to spread LDS banks
#define BSTRIDE (KC + BPAD)  // 80 bf16 = 160 B row stride

typedef __attribute__((ext_vector_type(16))) __bf16 v16bf;
typedef __attribute__((ext_vector_type(8)))  float  v8f;

// ---------------------------------------------------------------- prep: q = h_t @ w_h^T
__global__ __launch_bounds__(256) void prep_q_kernel(
    const float* __restrict__ h_t, const float* __restrict__ w_h, float* __restrict__ q)
{
    int b = blockIdx.x;          // 0..63
    int a = threadIdx.x;         // 0..255
    const float* hp = h_t + (size_t)b * HID_;
    const float* wp = w_h + (size_t)a * HID_;
    float s = 0.f;
    #pragma unroll 8
    for (int h = 0; h < HID_; ++h) s = fmaf(hp[h], wp[h], s);
    q[b * ATTN_ + a] = s;
}

// ---------------------------------------------------------------- prep: w_m f32 -> bf16
__global__ __launch_bounds__(256) void prep_wm_kernel(
    const float* __restrict__ w_m, __bf16* __restrict__ wmB)
{
    int i = blockIdx.x * blockDim.x + threadIdx.x;   // grid covers 256*1024
    wmB[i] = (__bf16)w_m[i];
}

// ---------------------------------------------------------------- prep: zero accumulators
__global__ __launch_bounds__(256) void zero_ws_kernel(
    float* __restrict__ ctx, float* __restrict__ S)
{
    int i = blockIdx.x * blockDim.x + threadIdx.x;
    if (i < B_ * MEM_) ctx[i] = 0.f;
    if (i < B_)        S[i]   = 0.f;
}

// ---------------------------------------------------------------- main fused kernel
// grid = (B, T/TM), block = 256 threads (8 waves of 32)
__global__ __launch_bounds__(256) void attn_main_kernel(
    const float*         __restrict__ memory,   // [B,T,MEM] f32
    const unsigned char* __restrict__ mask,     // [B,T] bool
    const __bf16*        __restrict__ wmB,      // [ATTN,MEM] bf16
    const float*         __restrict__ q,        // [B,ATTN]
    const float*         __restrict__ v,        // [ATTN]
    float*               __restrict__ wOut,     // [B,T] unnormalized exp weights
    float*               __restrict__ ctx,      // [B,MEM] accum
    float*               __restrict__ S)        // [B] accum
{
    __shared__ __bf16 sB[ATTN_ * BSTRIDE];      // 40 KB staged w_m chunk
    __shared__ float  wRow[TM];

    const int b    = blockIdx.x;
    const int t0   = blockIdx.y * TM;
    const int tid  = threadIdx.x;
    const int wv   = tid >> 5;                  // wave 0..7 (wave32)
    const int lane = tid & 31;
    const int l15  = lane & 15;
    const int hv   = lane >> 4;                 // lane half (K-layout split)

    const float* memB   = memory + ((size_t)b * T_ + t0) * MEM_;
    const int    r      = wv * 16 + l15;        // this lane's A-matrix row
    const float* rowPtr = memB + (size_t)r * MEM_;

    v8f acc[16];
    {
        v8f z = {0.f,0.f,0.f,0.f,0.f,0.f,0.f,0.f};
        #pragma unroll
        for (int i = 0; i < 16; ++i) acc[i] = z;
    }

    for (int kc = 0; kc < MEM_; kc += KC) {
        // stage B chunk: thread tid copies row n=tid, K [kc, kc+KC)
        {
            const uint4* src = (const uint4*)(wmB + (size_t)tid * MEM_ + kc);
            uint4*       dst = (uint4*)(sB + tid * BSTRIDE);
            #pragma unroll
            for (int kk = 0; kk < KC / 8; ++kk) dst[kk] = src[kk];
        }
        __syncthreads();

        #pragma unroll
        for (int kq = 0; kq < KC; kq += 32) {
            const int k0 = kc + kq;
            const int ks = hv * 8;
            // A fragment: lane holds K [hv*16, hv*16+16) contiguous (this code's k order, same as B)
            const float4* pa = (const float4*)(rowPtr + k0 + hv * 16);
            const float4* pb = (const float4*)(rowPtr + k0 + hv * 16 + 8);
            float4 a0 = pa[0], a1 = pa[1], a2 = pb[0], a3 = pb[1];
            v16bf af;
            af[0]=(__bf16)a0.x; af[1]=(__bf16)a0.y; af[2]=(__bf16)a0.z; af[3]=(__bf16)a0.w;
            af[4]=(__bf16)a1.x; af[5]=(__bf16)a1.y; af[6]=(__bf16)a1.z; af[7]=(__bf16)a1.w;
            af[8]=(__bf16)a2.x; af[9]=(__bf16)a2.y; af[10]=(__bf16)a2.z; af[11]=(__bf16)a2.w;
            af[12]=(__bf16)a3.x; af[13]=(__bf16)a3.y; af[14]=(__bf16)a3.z; af[15]=(__bf16)a3.w;

            #pragma unroll
            for (int ct = 0; ct < 16; ++ct) {
                // B fragment: lane = col n, 16 contiguous K per lane (same k order as A)
                const int n = ct * 16 + l15;
                v16bf bf = *(const v16bf*)(sB + n * BSTRIDE + kq + hv * 16);
                acc[ct] = __builtin_amdgcn_wmma_f32_16x16x32_bf16(
                    false, af, false, bf, (short)0, acc[ct], false, false);
            }
        }
        __syncthreads();
    }

    // e[t] = sum_n v[n] * tanh(q[b][n] + Kproj[t][n])
    // C layout: vgpr i -> row M = i + hv*8 ; col N = l15 (per tile)
    float part[8];
    #pragma unroll
    for (int i = 0; i < 8; ++i) part[i] = 0.f;

    const float* qb = q + b * ATTN_;
    #pragma unroll
    for (int ct = 0; ct < 16; ++ct) {
        const int n = ct * 16 + l15;
        const float vn = v[n];
        const float qn = qb[n];
        #pragma unroll
        for (int i = 0; i < 8; ++i)
            part[i] += vn * tanhf(qn + acc[ct][i]);
    }
    // reduce across the 16 lanes of each half (masks < 16 stay within half)
    #pragma unroll
    for (int m = 1; m < 16; m <<= 1) {
        #pragma unroll
        for (int i = 0; i < 8; ++i)
            part[i] += __shfl_xor(part[i], m, 32);
    }
    if (l15 == 0) {
        #pragma unroll
        for (int i = 0; i < 8; ++i) {
            const int trel = wv * 16 + hv * 8 + i;
            const int t    = t0 + trel;
            // |e| <= 16 so exp never overflows: max-free one-pass softmax
            const float w = mask[(size_t)b * T_ + t] ? __expf(part[i]) : 0.f;
            wOut[(size_t)b * T_ + t] = w;
            wRow[trel] = w;
        }
    }
    __syncthreads();

    if (tid == 0) {
        float s = 0.f;
        #pragma unroll 8
        for (int t = 0; t < TM; ++t) s += wRow[t];
        atomicAdd(S + b, s);
    }

    // fused context partial: tile was just streamed -> second read hits L2
    #pragma unroll
    for (int c = 0; c < MEM_ / 256; ++c) {
        const int m = tid + c * 256;
        float s = 0.f;
        #pragma unroll 4
        for (int t = 0; t < TM; ++t)
            s = fmaf(wRow[t], memB[(size_t)t * MEM_ + m], s);
        atomicAdd(ctx + b * MEM_ + m, s);
    }
}

// ---------------------------------------------------------------- finalize: normalize
__global__ __launch_bounds__(256) void finalize_kernel(
    const float* __restrict__ ctx, const float* __restrict__ wUn,
    const float* __restrict__ S, float* __restrict__ out)
{
    int i = blockIdx.x * blockDim.x + threadIdx.x;  // grid covers B*T
    if (i < B_ * MEM_) {
        int b = i >> 10;
        out[i] = ctx[i] / S[b];                     // context first in d_out
    }
    if (i < B_ * T_) {
        int b = i >> 12;
        out[B_ * MEM_ + i] = wUn[i] / S[b];         // then attn
    }
}

// ---------------------------------------------------------------- launch
extern "C" void kernel_launch(void* const* d_in, const int* in_sizes, int n_in,
                              void* d_out, int out_size, void* d_ws, size_t ws_size,
                              hipStream_t stream) {
    (void)in_sizes; (void)n_in; (void)out_size; (void)ws_size;
    const float*         h_t    = (const float*)d_in[0];
    const float*         memory = (const float*)d_in[1];
    const unsigned char* mmask  = (const unsigned char*)d_in[2];
    const float*         w_h    = (const float*)d_in[3];
    const float*         w_m    = (const float*)d_in[4];
    const float*         v      = (const float*)d_in[5];

    char*   ws  = (char*)d_ws;
    float*  q   = (float*)(ws + 0);                    // 64*256*4   = 64 KB
    __bf16* wmB = (__bf16*)(ws + 65536);               // 256*1024*2 = 512 KB
    float*  wUn = (float*)(ws + 589824);               // 64*4096*4  = 1 MB
    float*  ctx = (float*)(ws + 1638400);              // 64*1024*4  = 256 KB
    float*  S   = (float*)(ws + 1900544);              // 64*4
    float*  out = (float*)d_out;

    prep_q_kernel <<<B_, 256, 0, stream>>>(h_t, w_h, q);
    prep_wm_kernel<<<(ATTN_ * MEM_) / 256, 256, 0, stream>>>(w_m, wmB);
    zero_ws_kernel<<<(B_ * MEM_ + 255) / 256, 256, 0, stream>>>(ctx, S);

    dim3 grid(B_, T_ / TM);
    attn_main_kernel<<<grid, 256, 0, stream>>>(memory, mmask, wmB, q, v, wUn, ctx, S);

    finalize_kernel<<<(B_ * T_) / 256, 256, 0, stream>>>(ctx, wUn, S, out);
}
